// SplineFlow_49512382988356
// MI455X (gfx1250) — matrix operand order repaired and checked by reading, not compiled, not run on hardware
//
#include <hip/hip_runtime.h>

// SplineFlow (MADE + rational-quadratic spline), fully fused for gfx1250.
// B=32768, D=64, H1=64, H2=32, K=8 bins, M=3K-1=23, OUT=M*D=1472.
//
// L3 weight block (92KB, f16-packed) is DMA'd global->LDS once per block with
// the Tensor Data Mover (tensor_load_to_lds, 6-arg clang-23 form), overlapped
// with the L1/L2 GEMMs; the 92 WMMA B-operand loads per wave come from LDS.

#define B_ROWS  32768
#define D_DIM   64
#define H1_DIM  64
#define H2_DIM  32
#define KBINS   8
#define M_PAR   23
#define OUT_F   1472

typedef __attribute__((ext_vector_type(16))) _Float16 v16h;
typedef __attribute__((ext_vector_type(8)))  float    v8f;
typedef __attribute__((ext_vector_type(4)))  unsigned int v4u;
typedef __attribute__((ext_vector_type(8)))  int      v8i;
typedef __attribute__((ext_vector_type(4)))  int      v4i;

// f16-packed masked weights in workspace (units: halfs)
#define BW1_OFF 0        // 8 tiles  * 512 = 4096
#define BW2_OFF 4096     // 4 tiles  * 512 = 2048
#define BW3_OFF 6144     // 92 tiles * 512 = 47104
#define BW3_HALFS (92 * 512)
#define BW3_QWORDS (BW3_HALFS / 4)   // 11776 x 8B
#define TOT_HALFS ((8 + 4 + 92) * 512)

// -------------------------------------------------------------------------
// Pack masked weights into WMMA B-operand lane layout (f16).
// B-matrix (K x 16) per ISA: lane l -> column n = l&15; the 16 halfs of the
// lane hold K = 8*(l>>4) + hh (hh < 8), 16 + 8*(l>>4) + (hh - 8) (hh >= 8) for a 32-deep K chunk.
// MADE masks (identity permutation, D=64, H1=64, H2=32):
//   m1[j][d]  = (j%63) >= d ;  m2[j2][h] = j2 >= (h%63) ;  m3[o][i] = (o%64) > i
// -------------------------------------------------------------------------
__global__ void prep_weights(const float* __restrict__ W1,
                             const float* __restrict__ W2,
                             const float* __restrict__ W3,
                             _Float16* __restrict__ ws) {
    int i = blockIdx.x * blockDim.x + threadIdx.x;
    if (i >= TOT_HALFS) return;
    if (i < 8 * 512) {                       // bw1: tiles (nt 0..3) x (kc 0..1)
        int t = i >> 9, r = i & 511;
        int l = r >> 4, hh = r & 15;
        int nt = t >> 1, kc = t & 1;
        int j = nt * 16 + (l & 15);           // output feature 0..63
        int k = kc * 32 + (hh < 8 ? 8 * (l >> 4) + hh : 16 + 8 * (l >> 4) + (hh - 8)); // input dim d 0..63
        float m = ((j % 63) >= k) ? 1.0f : 0.0f;
        ws[BW1_OFF + i] = (_Float16)(W1[j * D_DIM + k] * m);
    } else if (i < 12 * 512) {               // bw2: tiles (nt 0..1) x (kc 0..1)
        int ii = i - 8 * 512;
        int t = ii >> 9, r = ii & 511;
        int l = r >> 4, hh = r & 15;
        int nt = t >> 1, kc = t & 1;
        int j = nt * 16 + (l & 15);           // 0..31
        int k = kc * 32 + (hh < 8 ? 8 * (l >> 4) + hh : 16 + 8 * (l >> 4) + (hh - 8)); // 0..63
        float m = (j >= (k % 63)) ? 1.0f : 0.0f;
        ws[BW2_OFF + ii] = (_Float16)(W2[j * H1_DIM + k] * m);
    } else {                                 // bw3: 92 tiles over OUT_F
        int ii = i - 12 * 512;
        int t = ii >> 9, r = ii & 511;
        int l = r >> 4, hh = r & 15;
        int o = t * 16 + (l & 15);            // 0..1471
        int k = (hh < 8 ? 8 * (l >> 4) + hh : 16 + 8 * (l >> 4) + (hh - 8)); // 0..31
        float m = ((o & 63) > k) ? 1.0f : 0.0f;
        ws[BW3_OFF + ii] = (_Float16)(W3[o * H2_DIM + k] * m);
    }
}

// -------------------------------------------------------------------------
// Fused forward: per wave, one 16-row batch tile; 4 waves per block.
// -------------------------------------------------------------------------
__launch_bounds__(128)
__global__ void spline_flow_kernel(const float* __restrict__ X,
                                   const float* __restrict__ b1,
                                   const float* __restrict__ b2,
                                   const float* __restrict__ b3,
                                   const _Float16* __restrict__ ws,
                                   float* __restrict__ out) {
    __shared__ _Float16 h1s[4][16 * H1_DIM];
    __shared__ _Float16 h2s[4][16 * H2_DIM];
    __shared__ alignas(16) _Float16 bw3s[BW3_HALFS];   // 94208 B, TDM destination

    const int tid   = threadIdx.x;
    const int l     = tid & 31;           // lane in wave32
    const int wv    = tid >> 5;           // wave in block (0..3)
    const int rl    = l & 15;             // row/col index within half-group
    const int base0 = (l >> 4) * 8;       // A-layout K offset / C-layout row offset
    const int row0  = blockIdx.x * 64 + wv * 16;

    const _Float16* bw1 = ws + BW1_OFF;
    const _Float16* bw2 = ws + BW2_OFF;
    _Float16* h1w = &h1s[wv][0];
    _Float16* h2w = &h2s[wv][0];

    // ---- Kick off TDM: bw3 (92KB) global -> LDS, one descriptor per block ----
    if (wv == 0) {
        unsigned long long ga = (unsigned long long)(uintptr_t)(ws + BW3_OFF);
        unsigned int lds_addr = (unsigned int)(uintptr_t)(&bw3s[0]);
        v4u g0;
        g0[0] = 1u;                                     // count=1, no gather
        g0[1] = lds_addr;                               // lds_addr [63:32]
        g0[2] = (unsigned int)(ga & 0xFFFFFFFFu);       // global_addr [95:64]
        g0[3] = (unsigned int)((ga >> 32) & 0x1FFFFFFu) // global_addr [120:96]
              | (2u << 30);                             // type=2 ("image")
        v8i g1 = {};
        g1[0] = (3 << 16);                              // data_size = 8B
        g1[1] = (int)((BW3_QWORDS & 0xFFFF) << 16);     // tensor_dim0[15:0] @48
        g1[2] = (int)((BW3_QWORDS >> 16) & 0xFFFF)      // tensor_dim0[31:16] @64
              | (1 << 16);                              // tensor_dim1 = 1 @80
        g1[3] = (int)((BW3_QWORDS & 0xFFFF) << 16);     // tile_dim0 @112
        g1[4] = 1;                                      // tile_dim1 = 1 @128
        g1[5] = (int)BW3_QWORDS;                        // tensor_dim0_stride @160
        g1[6] = 0;
        g1[7] = 0;
        v4i gz4 = {};
        v8i gz8 = {};
        __builtin_amdgcn_tensor_load_to_lds(g0, g1, gz4, gz4, gz8, 0);
    }

    // ---------------- Layer 1: h1 = relu(x @ (W1*m1)^T + b1) ----------------
    v16h a1[2];
#pragma unroll
    for (int kc = 0; kc < 2; ++kc) {
        const float* xr = X + (size_t)(row0 + rl) * D_DIM + kc * 32 + base0;
#pragma unroll
        for (int j = 0; j < 8; ++j) a1[kc][j] = (_Float16)xr[j];
#pragma unroll
        for (int j = 0; j < 8; ++j) a1[kc][8 + j] = (_Float16)xr[16 + j];
    }
#pragma unroll
    for (int nt = 0; nt < 4; ++nt) {
        v8f acc = {};
#pragma unroll
        for (int kc = 0; kc < 2; ++kc) {
            v16h b = *(const v16h*)(bw1 + ((nt * 2 + kc) * 32 + l) * 16);
            acc = __builtin_amdgcn_wmma_f32_16x16x32_f16(false, a1[kc], false, b,
                                                         (short)0, acc, false, false);
        }
        float bias = b1[nt * 16 + rl];
#pragma unroll
        for (int v = 0; v < 8; ++v) {
            float h = acc[v] + bias;           // C layout: row = v+base0, col = nt*16+rl
            h = h > 0.0f ? h : 0.0f;
            h1w[(v + base0) * H1_DIM + nt * 16 + rl] = (_Float16)h;
        }
    }

    // ---------------- Layer 2: h2 = relu(h1 @ (W2*m2)^T + b2) ----------------
    v16h a2[2];
#pragma unroll
    for (int kc = 0; kc < 2; ++kc) {
        const _Float16* hr = h1w + rl * H1_DIM + kc * 32 + base0;
#pragma unroll
        for (int j = 0; j < 8; ++j) a2[kc][j] = hr[j];
#pragma unroll
        for (int j = 0; j < 8; ++j) a2[kc][8 + j] = hr[16 + j];
    }
#pragma unroll
    for (int nt = 0; nt < 2; ++nt) {
        v8f acc = {};
#pragma unroll
        for (int kc = 0; kc < 2; ++kc) {
            v16h b = *(const v16h*)(bw2 + ((nt * 2 + kc) * 32 + l) * 16);
            acc = __builtin_amdgcn_wmma_f32_16x16x32_f16(false, a2[kc], false, b,
                                                         (short)0, acc, false, false);
        }
        float bias = b2[nt * 16 + rl];
#pragma unroll
        for (int v = 0; v < 8; ++v) {
            float h = acc[v] + bias;
            h = h > 0.0f ? h : 0.0f;
            h2w[(v + base0) * H2_DIM + nt * 16 + rl] = (_Float16)h;
        }
    }

    // ---------------- Layer 3 A operand (16x32, single K chunk) --------------
    v16h a3;
    {
        const _Float16* hr = h2w + rl * H2_DIM + base0;
#pragma unroll
        for (int j = 0; j < 8; ++j) a3[j] = hr[j];
#pragma unroll
        for (int j = 0; j < 8; ++j) a3[8 + j] = hr[16 + j];
    }

    // TDM must be complete (and visible to all waves) before L3 B reads.
    if (wv == 0) __builtin_amdgcn_s_wait_tensorcnt(0);
    __syncthreads();

    float ldacc[8];
#pragma unroll
    for (int v = 0; v < 8; ++v) ldacc[v] = 0.0f;

    // ---------------- Layer 3 + RQ spline, per 16-dim group ------------------
    for (int g = 0; g < 4; ++g) {
        const int d = g * 16 + rl;            // this lane's dim (C layout column)
        v8f acc[M_PAR];
#pragma unroll
        for (int m = 0; m < M_PAR; ++m) {
            v16h b = *(const v16h*)(&bw3s[((m * 4 + g) * 32 + l) * 16]);
            v8f c = {};
            acc[m] = __builtin_amdgcn_wmma_f32_16x16x32_f16(false, a3, false, b,
                                                            (short)0, c, false, false);
        }
        float b3v[M_PAR];
#pragma unroll
        for (int m = 0; m < M_PAR; ++m) b3v[m] = b3[m * D_DIM + d];

#pragma unroll
        for (int v = 0; v < 8; ++v) {
            const int row = row0 + v + base0;
            float P[M_PAR];
#pragma unroll
            for (int m = 0; m < M_PAR; ++m) P[m] = acc[m][v] + b3v[m];

            const float xv = X[(size_t)row * D_DIM + d];

            // widths softmax -> cumwidths on [-3,3]
            float mx = P[0];
#pragma unroll
            for (int k = 1; k < 8; ++k) mx = fmaxf(mx, P[k]);
            float ew[8], sw = 0.0f;
#pragma unroll
            for (int k = 0; k < 8; ++k) { ew[k] = expf(P[k] - mx); sw += ew[k]; }
            float invw = 1.0f / sw;
            float cumw[9], wd[8];
            cumw[0] = -3.0f;
            float cs = 0.0f;
#pragma unroll
            for (int k = 0; k < 7; ++k) {
                cs += 1e-3f + 0.992f * ew[k] * invw;   // MIN_BIN_WIDTH + (1-8e-3)*w
                cumw[k + 1] = -3.0f + 6.0f * cs;
            }
            cumw[8] = 3.0f;
#pragma unroll
            for (int k = 0; k < 8; ++k) wd[k] = cumw[k + 1] - cumw[k];

            // heights softmax -> cumheights
            float mh = P[8];
#pragma unroll
            for (int k = 1; k < 8; ++k) mh = fmaxf(mh, P[8 + k]);
            float eh[8], sh = 0.0f;
#pragma unroll
            for (int k = 0; k < 8; ++k) { eh[k] = expf(P[8 + k] - mh); sh += eh[k]; }
            float invh = 1.0f / sh;
            float cumh[9], ht[8];
            cumh[0] = -3.0f;
            float csh = 0.0f;
#pragma unroll
            for (int k = 0; k < 7; ++k) {
                csh += 1e-3f + 0.992f * eh[k] * invh;
                cumh[k + 1] = -3.0f + 6.0f * csh;
            }
            cumh[8] = 3.0f;
#pragma unroll
            for (int k = 0; k < 8; ++k) ht[k] = cumh[k + 1] - cumh[k];

            // derivatives: [1, 1e-3+softplus(P[16..22]), 1]
            float dv[9];
            dv[0] = 1.0f; dv[8] = 1.0f;
#pragma unroll
            for (int k = 0; k < 7; ++k) {
                float p = P[16 + k];
                float sp = p > 0.0f ? p + log1pf(expf(-p)) : log1pf(expf(p));
                dv[k + 1] = 1e-3f + sp;
            }

            const bool inside = (xv >= -3.0f) && (xv <= 3.0f);
            float xc = fminf(fmaxf(xv, -3.0f), 3.0f);
            int idx = -1;
#pragma unroll
            for (int k = 0; k < 9; ++k) idx += (xc >= cumw[k] + 1e-6f) ? 1 : 0;
            idx = idx < 0 ? 0 : (idx > 7 ? 7 : idx);

            float x_k = 0.f, w_k = 1.f, y_k = 0.f, h_k = 1.f, d_k = 1.f, d_k1 = 1.f;
#pragma unroll
            for (int k = 0; k < 8; ++k) {
                bool s = (idx == k);
                x_k  = s ? cumw[k]   : x_k;
                w_k  = s ? wd[k]     : w_k;
                y_k  = s ? cumh[k]   : y_k;
                h_k  = s ? ht[k]     : h_k;
                d_k  = s ? dv[k]     : d_k;
                d_k1 = s ? dv[k + 1] : d_k1;
            }

            float delta = h_k / w_k;
            float theta = (xc - x_k) / w_k;
            float omt = 1.0f - theta;
            float tt  = theta * omt;
            float th2 = theta * theta;
            float num = h_k * (delta * th2 + d_k * tt);
            float den = delta + (d_k + d_k1 - 2.0f * delta) * tt;
            float y   = y_k + num / den;
            float dnum = delta * delta * (d_k1 * th2 + 2.0f * delta * tt + d_k * omt * omt);
            float ld  = logf(dnum) - 2.0f * logf(den);

            out[(size_t)row * D_DIM + d] = inside ? y : xv;
            ldacc[v] += inside ? ld : 0.0f;
        }
    }

    // log-det reduction over the 16 dims of each half-wave group, all 4 groups
#pragma unroll
    for (int v = 0; v < 8; ++v) {
        float s = ldacc[v];
        s += __shfl_xor(s, 1, 32);
        s += __shfl_xor(s, 2, 32);
        s += __shfl_xor(s, 4, 32);
        s += __shfl_xor(s, 8, 32);
        if (rl == 0) out[(size_t)B_ROWS * D_DIM + row0 + v + base0] = s;
    }
}

extern "C" void kernel_launch(void* const* d_in, const int* in_sizes, int n_in,
                              void* d_out, int out_size, void* d_ws, size_t ws_size,
                              hipStream_t stream) {
    (void)in_sizes; (void)n_in; (void)out_size; (void)ws_size;
    const float* X  = (const float*)d_in[0];
    const float* W1 = (const float*)d_in[1];
    const float* b1 = (const float*)d_in[2];
    const float* W2 = (const float*)d_in[3];
    const float* b2 = (const float*)d_in[4];
    const float* W3 = (const float*)d_in[5];
    const float* b3 = (const float*)d_in[6];
    _Float16* ws = (_Float16*)d_ws;

    prep_weights<<<(TOT_HALFS + 255) / 256, 256, 0, stream>>>(W1, W2, W3, ws);
    spline_flow_kernel<<<B_ROWS / 64, 128, 0, stream>>>(X, b1, b2, b3, ws,
                                                        (float*)d_out);
}
